// FlashAttention_84404697301104
// MI455X (gfx1250) — matrix operand order repaired and checked by reading, not compiled, not run on hardware
//
#include <hip/hip_runtime.h>
#include <hip/hip_bf16.h>
#include <math.h>

// ---------------------------------------------------------------------------
// MI455X (gfx1250) fused multi-head attention:
//   Q = q@Wq^T+bq ; K = k@Wk^T+bk ; V = v@Wv^T+bv   (WMMA f16 GEMM, fp32 acc)
//   O = softmax(Q K^T / sqrt(64)) V                  (flash-style, WMMA, DPP softmax,
//                                                     TDM/async K-tile staging)
//   out = O@Wo^T+bo                                  (WMMA, fp32 stores)
// ---------------------------------------------------------------------------

typedef __attribute__((ext_vector_type(16))) _Float16 v16h;
typedef __attribute__((ext_vector_type(8)))  _Float16 v8h;
typedef __attribute__((ext_vector_type(8)))  float    v8f;
typedef __attribute__((ext_vector_type(4)))  unsigned int v4u;
typedef __attribute__((ext_vector_type(8)))  int      v8i;
typedef __attribute__((ext_vector_type(4)))  int      v4i;

#if defined(__has_builtin)
#if __has_builtin(__builtin_amdgcn_tensor_load_to_lds) && __has_builtin(__builtin_amdgcn_s_wait_tensorcnt)
#define USE_TDM 1
#endif
#endif
#ifndef USE_TDM
#define USE_TDM 0
#endif

constexpr int D_EMBED = 1024;
constexpr int N_HEADS = 16;
constexpr int HD      = 64;            // head dim
constexpr int SEQ     = 2048;
constexpr int BATCH   = 2;
constexpr int MTOT    = BATCH * SEQ;   // 4096 rows

static __device__ __forceinline__ v8f wmma_f16(v16h a, v16h b, v8f c) {
  // D = A(16x32 f16) x B(32x16 f16) + C(16x16 f32)
  return __builtin_amdgcn_wmma_f32_16x16x32_f16(
      /*neg_a=*/false, a, /*neg_b=*/false, b,
      /*c_mod=*/(short)0, c, /*reuse_a=*/false, /*reuse_b=*/false);
}

static __device__ __forceinline__ v16h cat8(v8h lo, v8h hi) {
  return __builtin_shufflevector(lo, hi, 0,1,2,3,4,5,6,7,8,9,10,11,12,13,14,15);
}
static __device__ __forceinline__ v8h cvt8(v8f x) {
  return __builtin_convertvector(x, v8h);
}

// ---- DPP16 cross-lane reduction within 16-lane groups (VALU, no LDS port) ----
template <int CTRL>
static __device__ __forceinline__ float dppmov(float x) {
  return __int_as_float(__builtin_amdgcn_update_dpp(
      __float_as_int(x), __float_as_int(x), CTRL, 0xF, 0xF, true));
}
static __device__ __forceinline__ float red_max16(float x) {
  x = fmaxf(x, dppmov<0xB1>(x));    // quad_perm(1,0,3,2)  : xor 1
  x = fmaxf(x, dppmov<0x4E>(x));    // quad_perm(2,3,0,1)  : xor 2
  x = fmaxf(x, dppmov<0x141>(x));   // row_half_mirror     : combine quads in 8
  x = fmaxf(x, dppmov<0x140>(x));   // row_mirror          : combine halves in 16
  return x;
}
static __device__ __forceinline__ float red_sum16(float x) {
  x += dppmov<0xB1>(x);
  x += dppmov<0x4E>(x);
  x += dppmov<0x141>(x);
  x += dppmov<0x140>(x);
  return x;
}

// A-fragment (16x32, 16-bit): lanes 0-15 row=lane, 16 contiguous k per lane;
// lanes 16-31 same rows, +16 half offset.  p already includes the +0/+16 lane offset.
static __device__ __forceinline__ v16h load_a_frag(const float* p) {
  v8f lo = *(const v8f*)p;
  v8f hi = *(const v8f*)(p + 8);
  return cat8(cvt8(lo), cvt8(hi));
}
static __device__ __forceinline__ v16h load_a_frag(const _Float16* p) {
  v8h lo = *(const v8h*)p;
  v8h hi = *(const v8h*)(p + 8);
  return cat8(lo, hi);
}

// B-fragment (32x16, 16-bit): lane n = col, 16 contiguous k per lane,
// lanes 16-31 offset +16 halves.  p already includes the +0/+16 lane offset.
static __device__ __forceinline__ v16h load_b_frag(const float* p) {
  v8f lo = *(const v8f*)p;
  v8f hi = *(const v8f*)(p + 8);
  return cat8(cvt8(lo), cvt8(hi));
}
static __device__ __forceinline__ v16h load_b_frag(const _Float16* p) {
  return *(const v16h*)p;
}

// ---------------------------------------------------------------------------
// elementwise fp32 -> f16 (vectorized x8)
// ---------------------------------------------------------------------------
__global__ __launch_bounds__(256)
void f32_to_f16_kernel(const float* __restrict__ src, _Float16* __restrict__ dst,
                       int n8)
{
  const int i = blockIdx.x * blockDim.x + threadIdx.x;
  if (i < n8) ((v8h*)dst)[i] = cvt8(((const v8f*)src)[i]);
}

// ---------------------------------------------------------------------------
// GEMM: C[m,n] = sum_k X[m,k] * W[n,k] + bias[n]
//   block = 256 threads = 8 waves; wave -> 16x64 strip; K stepped by 32.
// ---------------------------------------------------------------------------
template <typename TX, typename TW, bool OUT16>
__global__ __launch_bounds__(256)
void gemm_xwt_kernel(const TX* __restrict__ X, const TW* __restrict__ W,
                     const float* __restrict__ bias,
                     _Float16* __restrict__ out16, float* __restrict__ out32,
                     int M, int N, int K)
{
  const int lane = threadIdx.x & 31;
  const int wave = threadIdx.x >> 5;
  const int half = lane >> 4;
  const int l16  = lane & 15;

  const int m0 = blockIdx.y * 128 + wave * 16;
  const int n0 = blockIdx.x * 64;

  v8f acc[4];
#pragma unroll
  for (int f = 0; f < 4; ++f) {
    const float bv = bias[n0 + f * 16 + l16];
#pragma unroll
    for (int r = 0; r < 8; ++r) acc[f][r] = bv;
  }

  const size_t rowA = (size_t)(m0 + l16) * (size_t)K;
  const int aoff = half ? 16 : 0;
  const int boff = half ? 16 : 0;

  for (int k0 = 0; k0 < K; k0 += 32) {
    const v16h af = load_a_frag(X + rowA + k0 + aoff);
#pragma unroll
    for (int f = 0; f < 4; ++f) {
      // B[k][n] = W[n][k] : contiguous k along row n of W
      const TW* wp = W + (size_t)(n0 + f * 16 + l16) * (size_t)K + k0 + boff;
      acc[f] = wmma_f16(af, load_b_frag(wp), acc[f]);
    }
  }

#pragma unroll
  for (int f = 0; f < 4; ++f) {
    const int n = n0 + f * 16 + l16;
#pragma unroll
    for (int r = 0; r < 8; ++r) {
      const size_t m = (size_t)(m0 + r + 8 * half);
      if constexpr (OUT16) out16[m * (size_t)N + n] = (_Float16)acc[f][r];
      else                 out32[m * (size_t)N + n] = acc[f][r];
    }
  }
}

// ---------------------------------------------------------------------------
// Flash attention: one workgroup per (batch, head, 128-query block).
//   wave -> 16 query rows; key blocks of 64; online softmax in fp32 (DPP).
//   K tile staged via TDM (TENSORcnt) when available, else async loads.
// ---------------------------------------------------------------------------
__global__ __launch_bounds__(256)
void attention_kernel(const _Float16* __restrict__ Q,
                      const _Float16* __restrict__ K,
                      const _Float16* __restrict__ V,
                      _Float16* __restrict__ O)
{
  __shared__ alignas(16) _Float16 ksh[64][64];      // K block (key, hd)
  __shared__ alignas(16) _Float16 vsh[64][64];      // V block transposed (hd, key)
  __shared__ alignas(16) _Float16 psh[8][16][64];   // per-wave P relayout tile

  const int lane = threadIdx.x & 31;
  const int wave = threadIdx.x >> 5;
  const int half = lane >> 4;
  const int l16  = lane & 15;

  const int qblk = blockIdx.x;    // 0..15
  const int h    = blockIdx.y;    // 0..15
  const int b    = blockIdx.z;    // 0..1

  const int    D     = D_EMBED;
  const size_t hoff  = (size_t)h * HD;
  const size_t rowb  = (size_t)b * SEQ;
  const int    qbase = qblk * 128 + wave * 16;

  const int aoff = half ? 16 : 0;
  const int bofs = half ? 16 : 0;

  // Q A-fragments for k=0..31 / 32..63 (loaded once)
  const size_t qrow = (rowb + qbase + l16) * (size_t)D + hoff;
  v16h qa[2];
  qa[0] = cat8(*(const v8h*)(Q + qrow + 0  + aoff), *(const v8h*)(Q + qrow + 8  + aoff));
  qa[1] = cat8(*(const v8h*)(Q + qrow + 32 + aoff), *(const v8h*)(Q + qrow + 40 + aoff));

  v8f acc[4] = {};
  float mrow[8], lrow[8];
#pragma unroll
  for (int r = 0; r < 8; ++r) { mrow[r] = -INFINITY; lrow[r] = 0.0f; }

  // V cooperative-load indexing: 64x64 halves, 16 per thread
  const int vrow = threadIdx.x >> 2;         // 0..63 key row
  const int vcol = (threadIdx.x & 3) * 16;   // 0,16,32,48 head-dim col

  constexpr float SCALE = 0.125f;            // 1/sqrt(64)

  for (int kb = 0; kb < SEQ; kb += 64) {
    __syncthreads();  // all waves done reading tiles of previous iteration

#if USE_TDM
    // --- K tile: single Tensor Data Mover descriptor, issued by wave 0 ---
    if (wave == 0) {
      const unsigned lds_base = (unsigned)(size_t)&ksh[0][0];
      const unsigned long long ga =
          (unsigned long long)(size_t)(K + (rowb + kb) * (size_t)D + hoff);
      v4u g0;
      g0[0] = 1u;                                   // count=1, user mode, no gather
      g0[1] = lds_base;                             // lds_addr
      g0[2] = (unsigned)(ga & 0xFFFFFFFFu);         // global_addr[31:0]
      g0[3] = (unsigned)((ga >> 32) & 0x01FFFFFFu)  // global_addr[56:32]
              | (2u << 30);                         // type = 2 (image/tensor)
      v8i g1;
      g1[0] = (int)(1u << 16);                      // data_size=1 (2B); mask=0
      g1[1] = (int)(64u << 16);                     // tensor_dim0[15:0]=64
      g1[2] = (int)(64u << 16);                     // tensor_dim0 hi=0; tensor_dim1[15:0]=64
      g1[3] = (int)(64u << 16);                     // tensor_dim1 hi=0; tile_dim0=64
      g1[4] = (int)(64u);                           // tile_dim1=64; tile_dim2=0
      g1[5] = (int)(unsigned)D;                     // tensor_dim0_stride = 1024 elems
      g1[6] = 0;                                    // stride hi / dim1_stride lo
      g1[7] = 0;
      v4i z4 = {0, 0, 0, 0};                        // 2-D tile: groups 2/3 unused
      v8i z8 = {0, 0, 0, 0, 0, 0, 0, 0};            // trailing group (clang-23 form)
      __builtin_amdgcn_tensor_load_to_lds(g0, g1, z4, z4, z8, 0);
    }
#else
    // --- K tile: async global->LDS copy, 16B per lane, 2 chunks per wave ---
#pragma unroll
    for (int c = 0; c < 2; ++c) {
      const int byteoff = (wave * 2 + c) * 512 + lane * 16;  // within 8KB tile
      const int kr = byteoff >> 7;                            // 128B per row
      const int kc = (byteoff & 127) >> 1;                    // half index
      const unsigned ldsaddr = (unsigned)(size_t)&ksh[kr][kc];
      const unsigned long long ga =
          (unsigned long long)(size_t)(K + (rowb + kb + kr) * (size_t)D + hoff + kc);
      asm volatile("global_load_async_to_lds_b128 %0, %1, off"
                   :: "v"(ldsaddr), "v"(ga) : "memory");
    }
#endif

    // --- V tile: load rows, store transposed into vsh ---
    {
      const size_t grow = (rowb + kb + vrow) * (size_t)D + hoff + vcol;
      const v8h v0 = *(const v8h*)(V + grow);
      const v8h v1 = *(const v8h*)(V + grow + 8);
#pragma unroll
      for (int e = 0; e < 8; ++e) {
        vsh[vcol + e][vrow]     = v0[e];
        vsh[vcol + 8 + e][vrow] = v1[e];
      }
    }

#if USE_TDM
    __builtin_amdgcn_s_wait_tensorcnt((short)0);
#else
    asm volatile("s_wait_asynccnt 0x0" ::: "memory");
#endif
    __syncthreads();

    // --- scores: 16 queries x 64 keys as four 16x16 fragments ---
    v8f s[4] = {};
#pragma unroll
    for (int j = 0; j < 4; ++j) {
      const v16h b0 = *(const v16h*)&ksh[j * 16 + l16][0 + bofs];
      s[j] = wmma_f16(qa[0], b0, s[j]);
      const v16h b1 = *(const v16h*)&ksh[j * 16 + l16][32 + bofs];
      s[j] = wmma_f16(qa[1], b1, s[j]);
    }

    // --- online softmax per row (row r+8*half lives in VGPR r across 16 lanes)
#pragma unroll
    for (int r = 0; r < 8; ++r) {
      float smax = fmaxf(fmaxf(s[0][r], s[1][r]), fmaxf(s[2][r], s[3][r]));
      smax = red_max16(smax) * SCALE;
      const float mnew = fmaxf(mrow[r], smax);
      const float corr = __expf(mrow[r] - mnew);
      const float p0 = __expf(s[0][r] * SCALE - mnew);
      const float p1 = __expf(s[1][r] * SCALE - mnew);
      const float p2 = __expf(s[2][r] * SCALE - mnew);
      const float p3 = __expf(s[3][r] * SCALE - mnew);
      const float rs = red_sum16((p0 + p1) + (p2 + p3));
      lrow[r] = lrow[r] * corr + rs;
      mrow[r] = mnew;
      acc[0][r] *= corr; acc[1][r] *= corr;
      acc[2][r] *= corr; acc[3][r] *= corr;
      const int pr = r + 8 * half;
      psh[wave][pr][l16]      = (_Float16)p0;
      psh[wave][pr][16 + l16] = (_Float16)p1;
      psh[wave][pr][32 + l16] = (_Float16)p2;
      psh[wave][pr][48 + l16] = (_Float16)p3;
    }
    __syncthreads();  // cross-lane visibility of per-wave P tile

    // --- reload P in A-fragment layout, accumulate P(16x64) x V(64x64) ---
    const v16h pa0 = cat8(*(const v8h*)&psh[wave][l16][0  + aoff],
                          *(const v8h*)&psh[wave][l16][8  + aoff]);
    const v16h pa1 = cat8(*(const v8h*)&psh[wave][l16][32 + aoff],
                          *(const v8h*)&psh[wave][l16][40 + aoff]);
#pragma unroll
    for (int f = 0; f < 4; ++f) {
      const v16h vb0 = *(const v16h*)&vsh[f * 16 + l16][0 + bofs];
      acc[f] = wmma_f16(pa0, vb0, acc[f]);
      const v16h vb1 = *(const v16h*)&vsh[f * 16 + l16][32 + bofs];
      acc[f] = wmma_f16(pa1, vb1, acc[f]);
    }
  }

  // --- epilogue: normalize, store per-head output (f16, (B*S, D) layout) ---
#pragma unroll
  for (int r = 0; r < 8; ++r) {
    const float invl = 1.0f / lrow[r];
    const size_t m = rowb + qbase + r + 8 * half;
#pragma unroll
    for (int f = 0; f < 4; ++f) {
      O[m * (size_t)D + hoff + f * 16 + l16] = (_Float16)(acc[f][r] * invl);
    }
  }
}

// ---------------------------------------------------------------------------
extern "C" void kernel_launch(void* const* d_in, const int* in_sizes, int n_in,
                              void* d_out, int out_size, void* d_ws, size_t ws_size,
                              hipStream_t stream)
{
  (void)in_sizes; (void)n_in; (void)out_size;

  const float* query = (const float*)d_in[0];
  const float* key   = (const float*)d_in[1];
  const float* value = (const float*)d_in[2];
  const float* Wq = (const float*)d_in[3];
  const float* bq = (const float*)d_in[4];
  const float* Wk = (const float*)d_in[5];
  const float* bk = (const float*)d_in[6];
  const float* Wv = (const float*)d_in[7];
  const float* bv = (const float*)d_in[8];
  const float* Wo = (const float*)d_in[9];
  const float* bo = (const float*)d_in[10];

  const size_t elems  = (size_t)MTOT * D_EMBED;     // 4 Mi
  const size_t welems = (size_t)D_EMBED * D_EMBED;  // 1 Mi
  _Float16* Q16 = (_Float16*)d_ws;
  _Float16* K16 = Q16 + elems;
  _Float16* V16 = K16 + elems;
  _Float16* O16 = V16 + elems;                      // 32 MB

  const dim3 gg(D_EMBED / 64, MTOT / 128);          // (16, 32)
  const dim3 bb(256);
  const dim3 ga(SEQ / 128, N_HEADS, BATCH);

  const bool w16path = ws_size >= (4 * elems + 4 * welems) * sizeof(_Float16); // 40 MB
  if (w16path) {
    _Float16* Wq16 = O16 + elems;
    _Float16* Wk16 = Wq16 + welems;
    _Float16* Wv16 = Wk16 + welems;
    _Float16* Wo16 = Wv16 + welems;
    const int n8 = (int)(welems / 8);
    const int cg = (n8 + 255) / 256;
    f32_to_f16_kernel<<<cg, bb, 0, stream>>>(Wq, Wq16, n8);
    f32_to_f16_kernel<<<cg, bb, 0, stream>>>(Wk, Wk16, n8);
    f32_to_f16_kernel<<<cg, bb, 0, stream>>>(Wv, Wv16, n8);
    f32_to_f16_kernel<<<cg, bb, 0, stream>>>(Wo, Wo16, n8);

    gemm_xwt_kernel<float, _Float16, true><<<gg, bb, 0, stream>>>(
        query, Wq16, bq, Q16, nullptr, MTOT, D_EMBED, D_EMBED);
    gemm_xwt_kernel<float, _Float16, true><<<gg, bb, 0, stream>>>(
        key,   Wk16, bk, K16, nullptr, MTOT, D_EMBED, D_EMBED);
    gemm_xwt_kernel<float, _Float16, true><<<gg, bb, 0, stream>>>(
        value, Wv16, bv, V16, nullptr, MTOT, D_EMBED, D_EMBED);

    attention_kernel<<<ga, bb, 0, stream>>>(Q16, K16, V16, O16);

    gemm_xwt_kernel<_Float16, _Float16, false><<<gg, bb, 0, stream>>>(
        O16, Wo16, bo, nullptr, (float*)d_out, MTOT, D_EMBED, D_EMBED);
  } else {
    gemm_xwt_kernel<float, float, true><<<gg, bb, 0, stream>>>(
        query, Wq, bq, Q16, nullptr, MTOT, D_EMBED, D_EMBED);
    gemm_xwt_kernel<float, float, true><<<gg, bb, 0, stream>>>(
        key,   Wk, bk, K16, nullptr, MTOT, D_EMBED, D_EMBED);
    gemm_xwt_kernel<float, float, true><<<gg, bb, 0, stream>>>(
        value, Wv, bv, V16, nullptr, MTOT, D_EMBED, D_EMBED);

    attention_kernel<<<ga, bb, 0, stream>>>(Q16, K16, V16, O16);

    gemm_xwt_kernel<_Float16, float, false><<<gg, bb, 0, stream>>>(
        O16, Wo, bo, nullptr, (float*)d_out, MTOT, D_EMBED, D_EMBED);
  }
}
